// BeamSearch_60653528154542
// MI455X (gfx1250) — compile-verified, matrix operand order checked by reading, not run on hardware
//
#include <hip/hip_runtime.h>

typedef __attribute__((ext_vector_type(16))) _Float16 v16h;
typedef __attribute__((ext_vector_type(8)))  float    v8f;

#define BB 64
#define NN 1024

// ---------------------------------------------------------------------------
// Kernel 1: per-(b,n) dynamic features + gating mask mc[b,n] (f16 -> d_ws)
// ---------------------------------------------------------------------------
__global__ void dynfeat_mask_kernel(const float* __restrict__ data,
                                    const float* __restrict__ current_time,
                                    const int*   __restrict__ current_poi_idx,
                                    const float* __restrict__ dist_mat,
                                    const float* __restrict__ mask,
                                    float*       __restrict__ dyn_feat,
                                    _Float16*    __restrict__ mc_out) {
    int idx = blockIdx.x * blockDim.x + threadIdx.x;
    if (idx >= BB * NN) return;
    int b = idx >> 10;            // / NN
    int n = idx & (NN - 1);

    const float* row = data + (size_t)idx * 8;
    float rise = row[0];
    float setw = row[1];
    float dur  = row[2];
    float arrv = row[3];
    float s2   = row[4];
    float s1   = row[5];
    float s0   = row[6];

    float tour_start = data[0];              // data[0,0,RISE]
    float max_dur    = data[3] - tour_start; // data[0,0,ARR] - tour_start

    float ct   = current_time[b];
    int   pidx = current_poi_idx[b];
    float arr  = ct + dist_mat[(size_t)pidx * NN + n];

    float poly = s2 * arr * arr + s1 * arr + s0;

    float* df = dyn_feat + (size_t)idx * 9;
    __builtin_nontemporal_store((ct   - rise)       / max_dur, df + 0);
    __builtin_nontemporal_store((setw - ct)         / max_dur, df + 1);
    __builtin_nontemporal_store((arrv - ct)         / max_dur, df + 2);
    __builtin_nontemporal_store((ct   - tour_start) / max_dur, df + 3);
    __builtin_nontemporal_store((arr  - tour_start) / max_dur, df + 4);
    __builtin_nontemporal_store((arr  - rise)       / max_dur, df + 5);
    __builtin_nontemporal_store((setw - arr)        / max_dur, df + 6);
    __builtin_nontemporal_store((arrv - arr)        / max_dur, df + 7);
    __builtin_nontemporal_store(poly / 100.0f,                 df + 8);

    bool ok = (arr >= rise) && (arr + dur <= arrv) && (poly > 0.0f);
    float mc = mask[idx] * (ok ? 1.0f : 0.0f);
    mc_out[idx] = (_Float16)mc;   // re-read by kernel 2: keep regular-temporal
}

// ---------------------------------------------------------------------------
// Kernel 2: adj[b] = mc[b]^T * mc[b] (rank-1 via WMMA, K=0 slice only),
// diagonal zeroed. One wave = 8 consecutive 16x16 tiles along tj.
// Block (8 waves) covers a full contiguous 16x1024 (64 KB) output strip.
// adj is write-once (268 MB > 192 MB L2) -> nontemporal stores.
// ---------------------------------------------------------------------------
__global__ void adj_wmma_kernel(const _Float16* __restrict__ mc,
                                float*          __restrict__ adj) {
    const int lane = threadIdx.x & 31;
    const int wib  = threadIdx.x >> 5;                 // wave in block (0..7)
    const int wid  = blockIdx.x * 8 + wib;             // global wave id (0..32767)

    const int b   = wid >> 9;          // 512 waves per batch (64 ti * 8 groups)
    const int rem = wid & 511;
    const int ti  = rem >> 3;          // row-tile index 0..63
    const int tjg = rem & 7;           // col-tile group -> tj in [8*tjg, 8*tjg+8)

    const int  n15 = lane & 15;
    const bool lo  = lane < 16;

    // A-matrix: 16x32 f16, only K=0 column populated.
    // ISA layout: lanes 0-15, VGPR0 low half = element K=0 of row M=lane.
    _Float16 av = mc[(size_t)b * NN + ti * 16 + n15];
    if (!lo) av = (_Float16)0.0f;
    v16h a = {};
    a[0] = av;

    const size_t row_base = ((size_t)b * NN + (size_t)ti * 16) * NN;
    const int mbase = lo ? 0 : 8;

    for (int t = 0; t < 8; ++t) {
        const int tj = tjg * 8 + t;

        // B-matrix: 32x16 f16, only K=0 row populated.
        // ISA layout: lanes 0-15, VGPR0 low half = element K=0 of col N=lane.
        _Float16 bv = mc[(size_t)b * NN + tj * 16 + n15];
        if (!lo) bv = (_Float16)0.0f;
        v16h bm = {};
        bm[0] = bv;

        v8f c = {};
        c = __builtin_amdgcn_wmma_f32_16x16x32_f16(
                /*neg_a=*/false, a, /*neg_b=*/false, bm,
                /*c_mod=*/(short)0, c, /*reuse_a=*/false, /*reuse_b=*/false);

        // D layout: VGPR v, lane L -> row M = v + 8*(L>=16), col N = L&15
        float* p = adj + row_base + (size_t)mbase * NN + (size_t)tj * 16 + n15;
        const bool diag_tile = (ti == tj);
#pragma unroll
        for (int v = 0; v < 8; ++v) {
            float x = c[v];
            if (diag_tile && (mbase + v) == n15) x = 0.0f;  // zero diagonal
            __builtin_nontemporal_store(x, p + (size_t)v * NN);
        }
    }
}

// ---------------------------------------------------------------------------
extern "C" void kernel_launch(void* const* d_in, const int* in_sizes, int n_in,
                              void* d_out, int out_size, void* d_ws, size_t ws_size,
                              hipStream_t stream) {
    const float* data         = (const float*)d_in[0];  // [B,N,8]
    const float* current_time = (const float*)d_in[1];  // [B,1]
    const int*   cur_poi      = (const int*)  d_in[2];  // [B]
    const float* dist_mat     = (const float*)d_in[3];  // [N,N]
    // d_in[4] = batch_idx (identity gather, unused)
    const float* mask         = (const float*)d_in[5];  // [B,N]

    float* out = (float*)d_out;
    float* dyn = out;                                   // [B,N,9]
    float* adj = out + (size_t)BB * NN * 9;             // [B,N,N]

    _Float16* mc = (_Float16*)d_ws;                     // [B,N] f16 gate

    // Phase 1: dynamic features + gate mask. 65536 threads.
    dynfeat_mask_kernel<<<(BB * NN) / 256, 256, 0, stream>>>(
        data, current_time, cur_poi, dist_mat, mask, dyn, mc);

    // Phase 2: 262144 16x16 tiles; 8 tiles/wave, 8 waves/block -> 4096 blocks.
    adj_wmma_kernel<<<4096, 256, 0, stream>>>(mc, adj);
}
